// RefLocal2_79998060855629
// MI455X (gfx1250) — matrix operand order repaired and checked by reading, not compiled, not run on hardware
//
#include <hip/hip_runtime.h>

typedef __attribute__((ext_vector_type(16))) _Float16 v16h;
typedef __attribute__((ext_vector_type(8)))  _Float16 v8h;
typedef __attribute__((ext_vector_type(4)))  _Float16 v4h;
typedef __attribute__((ext_vector_type(2)))  _Float16 v2h;
typedef __attribute__((ext_vector_type(8)))  float    v8f;
typedef __attribute__((ext_vector_type(4)))  float    v4f;

namespace {
constexpr int Bn = 4, H = 64, W = 64, C = 128;  // bin_size == C == 128
constexpr int TM  = 16;   // pixels per tile (WMMA M)
constexpr int NW  = 26;   // 25 neighbors + self
constexpr int WIN = 20;   // ref pixel window: 16 + 2*2
constexpr int DUMP = NW * TM;  // dump zone base inside sLog

__device__ __forceinline__ v8f wmma_f16(v16h a, v16h b, v8f c) {
  // D(f32 16x16) = A(f16 16x32) * B(f16 32x16) + C
  return __builtin_amdgcn_wmma_f32_16x16x32_f16(
      /*neg_a=*/false, a, /*neg_b=*/false, b,
      /*c_mod=*/(short)0, c, /*reuse_a=*/false, /*reuse_b=*/false);
}

// A-matrix fragment: lane m = lane&15, hi = lane>>4.
// e in 0..15 -> K = off + e, off = ck*32 + 16*hi.
__device__ __forceinline__ v16h make_a_frag(const _Float16* row, int off) {
  v8h lo = *(const v8h*)(row + off);
  v8h hh = *(const v8h*)(row + off + 8);
  v16h a;
#pragma unroll
  for (int e = 0; e < 8; ++e) { a[e] = lo[e]; a[e + 8] = hh[e]; }
  return a;
}

__device__ __forceinline__ int clampi(int x, int lo, int hi) {
  return x < lo ? lo : (x > hi ? hi : x);
}
} // namespace

__global__ __launch_bounds__(32)
void local_attn_wmma(const float* __restrict__ mainp,
                     const float* __restrict__ mainv,
                     const float* __restrict__ refp,
                     const float* __restrict__ refv,
                     float* __restrict__ out) {
  __shared__ alignas(32) _Float16 sMain[TM * C];    // main tile (f16)
  __shared__ alignas(32) _Float16 sRef[WIN * C];    // ref row window (f16)
  __shared__ alignas(32) _Float16 sValT[C * 32];    // value window, [c][k], k padded to 32
  __shared__ alignas(32) _Float16 sAband[TM * 32];  // banded attention A-matrix
  __shared__ alignas(32) _Float16 sAttn[NW * TM];   // attn weights (f16)
  __shared__ alignas(16) float    sLog[NW * TM + 32]; // logits/attn + 32-slot dump zone

  const int lane = threadIdx.x;
  const int m16  = lane & 15;
  const int hi   = lane >> 4;
  const int c4   = lane * 4;     // 4 consecutive channels per lane (128/4 == 32 lanes)

  const int w0 = blockIdx.x * TM;
  const int h  = blockIdx.y;
  const int b  = blockIdx.z;

  const size_t rowBase = ((size_t)(b * H + h)) * W * C;  // (b,h,0,0)

  // ---- stage main tile (f32 -> f16), vectorized & branch-free ----
#pragma unroll 4
  for (int m = 0; m < TM; ++m) {
    v4f v = *(const v4f*)&mainp[rowBase + (size_t)(w0 + m) * C + c4];
    v4h hv;
#pragma unroll
    for (int q = 0; q < 4; ++q) hv[q] = (_Float16)v[q];
    *(v4h*)&sMain[m * C + c4] = hv;
  }
  { // zero K-pad of sValT (k=20..31 never rewritten) and sAband (band refreshed per di)
    v8h z8 = {};
#pragma unroll
    for (int i = 0; i < 16; ++i) *(v8h*)&sValT[(i * 32 + lane) * 8] = z8;
#pragma unroll
    for (int i = 0; i < 2; ++i)  *(v8h*)&sAband[(i * 32 + lane) * 8] = z8;
  }
  __syncthreads();

  // A fragments of main tile: 4 K-chunks of 32
  v16h afr[4];
#pragma unroll
  for (int ck = 0; ck < 4; ++ck)
    afr[ck] = make_a_frag(&sMain[m16 * C], ck * 32 + hi * 16);

  // ================= Phase 1: attention logits via WMMA ==================
  for (int di = 0; di < 5; ++di) {
    const int hh = h + di - 2;
    const bool rOK = (hh >= 0) && (hh < H);
    const size_t rRow = ((size_t)(b * H + clampi(hh, 0, H - 1))) * W * C;
    // ref row window: clamped unconditional b128 loads, select-zero OOB
#pragma unroll 5
    for (int p = 0; p < WIN; ++p) {
      const int ww = w0 - 2 + p;
      const bool ok = rOK && (ww >= 0) && (ww < W);
      const int wc = clampi(ww, 0, W - 1);
      v4f v = *(const v4f*)&refp[rRow + (size_t)wc * C + c4];
      v4h hv;
#pragma unroll
      for (int q = 0; q < 4; ++q) hv[q] = (_Float16)(ok ? v[q] : 0.f);
      *(v4h*)&sRef[p * C + c4] = hv;
    }
    __syncthreads();

    v8f c0 = {}, c1 = {};
#pragma unroll
    for (int ck = 0; ck < 4; ++ck) {
      const int off = ck * 32 + hi * 16;                      // B: 16 contiguous K
      v16h b0 = *(const v16h*)(&sRef[(0 + m16) * C] + off);   // cols = win px 0..15
      c0 = wmma_f16(afr[ck], b0, c0);
      v16h b1 = *(const v16h*)(&sRef[(4 + m16) * C] + off);   // cols = win px 4..19
      c1 = wmma_f16(afr[ck], b1, c1);
    }
    // branch-free band extraction: off-band lanes store to the dump zone
#pragma unroll
    for (int r = 0; r < 8; ++r) {
      const int m  = r + 8 * hi;          // D(lane,r) holds (m, n=m16)
      const int d0 = m16 - m;             // tile0 diag -> dj+2 in [0,4]
      const int i0 = (d0 >= 0 && d0 <= 4) ? (di * 5 + d0) * TM + m : DUMP + lane;
      sLog[i0] = c0[r];
      const int d1 = d0 + 4;              // tile1 diag -> dj+2 in [0,4]
      const int i1 = (d1 >= 0 && d1 <= 4) ? (di * 5 + d1) * TM + m : DUMP + lane;
      sLog[i1] = c1[r];
    }
    __syncthreads();
  }
  { // self logit: diagonal of corr(main, main)
    v8f cs = {};
#pragma unroll
    for (int ck = 0; ck < 4; ++ck) {
      v16h bs = *(const v16h*)(&sMain[m16 * C] + ck * 32 + hi * 16);
      cs = wmma_f16(afr[ck], bs, cs);
    }
#pragma unroll
    for (int r = 0; r < 8; ++r) {
      const int m = r + 8 * hi;
      const int i = (m16 == m) ? 25 * TM + m : DUMP + lane;
      sLog[i] = cs[r];
    }
  }
  __syncthreads();

  // ================= Phase 2: softmax over 26 (lanes 0..15 = pixels) ====
  if (lane < TM) {
    float mx = -1e30f;
#pragma unroll
    for (int j = 0; j < NW; ++j) mx = fmaxf(mx, sLog[j * TM + lane]);
    float s = 0.f;
#pragma unroll
    for (int j = 0; j < NW; ++j) {
      const float e = __expf(sLog[j * TM + lane] - mx);
      s += e;
      sLog[j * TM + lane] = e;
    }
    const float inv = 1.f / s;
#pragma unroll
    for (int j = 0; j < NW; ++j) {
      const float a = sLog[j * TM + lane] * inv;
      sLog[j * TM + lane]  = a;            // keep f32 (self term)
      sAttn[j * TM + lane] = (_Float16)a;  // f16 for WMMA A-matrix
    }
  }
  __syncthreads();

  // ================= Phase 3: out = banded(attn) x values via WMMA ======
  v8f acc[8];
#pragma unroll
  for (int ct = 0; ct < 8; ++ct) { v8f z = {}; acc[ct] = z; }

  for (int di = 0; di < 5; ++di) {
    const int hh = h + di - 2;
    const bool rOK = (hh >= 0) && (hh < H);
    const size_t vRow = ((size_t)(b * H + clampi(hh, 0, H - 1))) * W * C;
    // value window transposed to [c][k]: paired k so LDS stores are b32
#pragma unroll 5
    for (int i = 0; i < WIN / 2; ++i) {
      const int k0 = 2 * i;
      const int ww0 = w0 - 2 + k0, ww1 = ww0 + 1;
      const bool ok0 = rOK && (ww0 >= 0) && (ww0 < W);
      const bool ok1 = rOK && (ww1 >= 0) && (ww1 < W);
      const int wc0 = clampi(ww0, 0, W - 1), wc1 = clampi(ww1, 0, W - 1);
      v4f a0 = *(const v4f*)&refv[vRow + (size_t)wc0 * C + c4];
      v4f a1 = *(const v4f*)&refv[vRow + (size_t)wc1 * C + c4];
#pragma unroll
      for (int q = 0; q < 4; ++q) {
        v2h p2;
        p2[0] = (_Float16)(ok0 ? a0[q] : 0.f);
        p2[1] = (_Float16)(ok1 ? a1[q] : 0.f);
        *(v2h*)&sValT[(c4 + q) * 32 + k0] = p2;  // k0 even -> 4B aligned
      }
    }
    // refresh band values (positions are di-invariant; padding stays zero)
    if (lane < TM) {
#pragma unroll
      for (int dj = 0; dj < 5; ++dj)    // k = m + dj  (window px = w0-2 + m + dj)
        sAband[lane * 32 + lane + dj] = sAttn[(di * 5 + dj) * TM + lane];
    }
    __syncthreads();

    v16h af = make_a_frag(&sAband[m16 * 32], hi * 16);
#pragma unroll
    for (int ct = 0; ct < 8; ++ct) {
      v16h bv = *(const v16h*)(&sValT[(ct * 16 + m16) * 32] + hi * 16);
      acc[ct] = wmma_f16(af, bv, acc[ct]);
    }
    __syncthreads();
  }

  // ---- fuse self term (f32, exact) and store D-layout accumulators ----
#pragma unroll
  for (int r = 0; r < 8; ++r) {
    const int m = r + 8 * hi;
    const float aself = sLog[25 * TM + m];
    const size_t base = rowBase + (size_t)(w0 + m) * C;
#pragma unroll
    for (int ct = 0; ct < 8; ++ct) {
      const int c = ct * 16 + m16;
      out[base + c] = acc[ct][r] + aself * mainv[base + c];
    }
  }
}

extern "C" void kernel_launch(void* const* d_in, const int* in_sizes, int n_in,
                              void* d_out, int out_size, void* d_ws, size_t ws_size,
                              hipStream_t stream) {
  (void)in_sizes; (void)n_in; (void)out_size; (void)d_ws; (void)ws_size;
  const float* mainp = (const float*)d_in[0];
  const float* mainv = (const float*)d_in[1];
  const float* refp  = (const float*)d_in[2];
  const float* refv  = (const float*)d_in[3];
  float* outp = (float*)d_out;

  dim3 grid(W / TM, H, Bn);   // (4, 64, 4) = 1024 single-wave workgroups
  dim3 block(32);             // one wave32 per tile
  local_attn_wmma<<<grid, block, 0, stream>>>(mainp, mainv, refp, refv, outp);
}
